// RCNN_49838800503575
// MI455X (gfx1250) — matrix operand order repaired and checked by reading, not compiled, not run on hardware
//
#include <hip/hip_runtime.h>

typedef __attribute__((ext_vector_type(16))) __bf16 v16bf;
typedef __attribute__((ext_vector_type(8)))  float  v8f;

__device__ __forceinline__ unsigned short f2bf(float f) {
  unsigned int u = __float_as_uint(f);
  u += 0x7FFFu + ((u >> 16) & 1u);           // round-to-nearest-even
  return (unsigned short)(u >> 16);
}
__device__ __forceinline__ float bf2f(unsigned short s) {
  return __uint_as_float(((unsigned int)s) << 16);
}

// ---------------------------------------------------------------------------
// Pack a [128,128,3,3] f32 weight tensor into WMMA B-fragment layout:
// dst[((nt*36 + t)*32 + lane)*8 + v] packs bf16 pair (k, k+1) for
//   t = (dy*3+dx)*4 + c0/32,  k = c0 + (lane>>4)*16 + 2v,  o = nt*16 + (lane&15)
// in the contiguous k order that conv3_wmma_lrn's A loads also use (V[v]: lanes0-15 K=2v,2v+1;
// lanes16-31 K=16+2v,16+2v+1; N = lane&15).
// ---------------------------------------------------------------------------
__global__ __launch_bounds__(256) void pack_w3x3(const float* __restrict__ src,
                                                 unsigned int* __restrict__ dst) {
  int i = blockIdx.x * 256 + threadIdx.x;          // 73728 total
  int v = i & 7;
  int r = i >> 3;
  int lane = r & 31;
  r >>= 5;
  int t  = r % 36;
  int nt = r / 36;
  int dy = t / 12;
  int rem = t % 12;
  int dx = rem >> 2;
  int c0 = (rem & 3) << 5;
  int c = c0 + ((lane >> 4) << 4) + (v << 1);
  int o = (nt << 4) + (lane & 15);
  unsigned int lo = f2bf(src[((o * 128 + c) * 3 + dy) * 3 + dx]);
  unsigned int hi = f2bf(src[((o * 128 + c + 1) * 3 + dy) * 3 + dx]);
  dst[i] = lo | (hi << 16);
}

// ---------------------------------------------------------------------------
// conv1 (3->128, 5x5, pad 2) per-channel sum / sumsq for batch-norm stats.
// Output is NOT stored; it is recomputed in conv1_pool (cheap vs 256MB store).
// ---------------------------------------------------------------------------
__global__ __launch_bounds__(256) void conv1_stats(const float* __restrict__ x,
                                                   const float* __restrict__ w1,
                                                   const float* __restrict__ b1,
                                                   float* __restrict__ stats) {
  __shared__ float rs[256], rq[256];
  int o = (int)(blockIdx.x & 127u);
  int pix = (int)(blockIdx.x >> 7) * 256 + (int)threadIdx.x;   // 0..524287
  int n = pix >> 12;
  int p = pix & 4095;
  int yy = p >> 6, xx = p & 63;
  const float* wo = w1 + o * 75;
  float acc = b1[o];
  for (int ci = 0; ci < 3; ++ci) {
    const float* xc = x + (((size_t)n * 3 + ci) << 12);
    for (int ky = 0; ky < 5; ++ky) {
      int iy = yy + ky - 2;
      if ((unsigned)iy >= 64u) continue;
      for (int kx = 0; kx < 5; ++kx) {
        int ix = xx + kx - 2;
        if ((unsigned)ix >= 64u) continue;
        acc += xc[(iy << 6) + ix] * wo[(ci * 5 + ky) * 5 + kx];
      }
    }
  }
  rs[threadIdx.x] = acc;
  rq[threadIdx.x] = acc * acc;
  __syncthreads();
  for (int s = 128; s > 0; s >>= 1) {
    if ((int)threadIdx.x < s) {
      rs[threadIdx.x] += rs[threadIdx.x + s];
      rq[threadIdx.x] += rq[threadIdx.x + s];
    }
    __syncthreads();
  }
  if (threadIdx.x == 0) {
    atomicAdd(&stats[o], rs[0]);
    atomicAdd(&stats[128 + o], rq[0]);
  }
}

__global__ void bn_finalize(const float* __restrict__ g, const float* __restrict__ b,
                            float* __restrict__ stats) {
  int c = threadIdx.x;                              // 128 threads
  const float invN = 1.0f / 524288.0f;
  float mean = stats[c] * invN;
  float var  = stats[128 + c] * invN - mean * mean;
  float sc = g[c] * rsqrtf(var + 1e-5f);
  stats[256 + c] = sc;
  stats[384 + c] = b[c] - mean * sc;
}

// ---------------------------------------------------------------------------
// Fused: recompute conv1, apply BN (precomputed scale/shift), ReLU,
// 3x3/s2/p1 maxpool 64->32. Output x2: NHWC bf16 [128,32,32,128].
// ---------------------------------------------------------------------------
__global__ __launch_bounds__(256) void conv1_pool(const float* __restrict__ x,
                                                  const float* __restrict__ w1,
                                                  const float* __restrict__ b1,
                                                  const float* __restrict__ stats,
                                                  unsigned short* __restrict__ x2) {
  unsigned int gid = blockIdx.x * 256u + threadIdx.x;   // 16,777,216
  int o = gid & 127;
  unsigned int r = gid >> 7;
  int ox = r & 31; r >>= 5;
  int oy = r & 31;
  int n  = r >> 5;
  float sc = stats[256 + o], sh = stats[384 + o];
  const float* wo = w1 + o * 75;
  float bo = b1[o];
  float m = -3.0e38f;
  for (int wy = 0; wy < 3; ++wy) {
    int iy0 = oy * 2 - 1 + wy;
    if ((unsigned)iy0 >= 64u) continue;
    for (int wx = 0; wx < 3; ++wx) {
      int ix0 = ox * 2 - 1 + wx;
      if ((unsigned)ix0 >= 64u) continue;
      float acc = bo;
      for (int ci = 0; ci < 3; ++ci) {
        const float* xc = x + (((size_t)n * 3 + ci) << 12);
        for (int ky = 0; ky < 5; ++ky) {
          int iy = iy0 + ky - 2;
          if ((unsigned)iy >= 64u) continue;
          for (int kx = 0; kx < 5; ++kx) {
            int ix = ix0 + kx - 2;
            if ((unsigned)ix >= 64u) continue;
            acc += xc[(iy << 6) + ix] * wo[(ci * 5 + ky) * 5 + kx];
          }
        }
      }
      m = fmaxf(m, fmaxf(acc * sc + sh, 0.0f));
    }
  }
  x2[gid] = f2bf(m);
}

// ---------------------------------------------------------------------------
// WMMA implicit-GEMM 3x3 conv (128->128, pad 1) fused with the RCL epilogue.
// Register-tiled: one workgroup (8 wave32) = 64 pixels (R full rows of width W)
// x 128 out channels. Each wave owns one 16-channel N-tile and keeps FOUR
// 16x16 accumulators (4 M-tiles), so each packed B fragment feeds 4
// v_wmma_f32_16x16x32_bf16 ops (4x B-traffic amortization). A streams from an
// LDS-staged zero-padded (R+2)x(W+2)x128 input window.
// fmode=1: s = conv(x)+bias, store s to fbuf.  fmode=0: s = conv(x)+fbuf.
// Then r = relu(s) / (sum_{|j-c|<=8} relu(s_j)^2 * 0.001/17 + 1)^0.75 -> bf16,
// with the banded LRN done through LDS (workgroup owns all 128 channels).
// ---------------------------------------------------------------------------
template <int W, int R>
__global__ __launch_bounds__(256) void conv3_wmma_lrn(
    const unsigned short* __restrict__ xin,   // NHWC bf16 [N,W,W,128]
    const unsigned int*  __restrict__ wp,     // packed weights [8][36][32][8]
    const float* __restrict__ bias,           // 128 (fmode=1) or nullptr
    float* __restrict__ fbuf,                 // NHWC f32 (write if fmode, else read)
    unsigned short* __restrict__ rout,        // NHWC bf16
    int fmode) {
  constexpr int H  = W;        // square maps only
  constexpr int WW = W + 2;    // padded window width
  constexpr int WR = R + 2;    // padded window rows
  __shared__ unsigned short sA[WR * WW * 128];  // input window, zero padded
  __shared__ float sval[R * W * 128];           // relu(s) for LRN (64x128)

  int bid = blockIdx.x;
  int yt = bid % (H / R);
  int n  = bid / (H / R);
  int y0 = yt * R;
  int tid = threadIdx.x;

  // Stage rows y0-1..y0+R, cols -1..W (u32 = 2 channels per load).
  constexpr int NW = WR * WW * 64;   // u32 words
  for (int i = tid; i < NW; i += 256) {
    int dy = i / (WW * 64);
    int rr = i - dy * (WW * 64);
    int xw = rr >> 6;
    int cp = rr & 63;
    int iy = y0 + dy - 1;
    int ix = xw - 1;
    unsigned int w = 0u;
    if ((unsigned)iy < (unsigned)H && (unsigned)ix < (unsigned)W)
      w = *(const unsigned int*)(xin + (((size_t)(n * H + iy) * W + ix) << 7) + (cp << 1));
    *(unsigned int*)(sA + (((dy * WW + xw) << 7) + (cp << 1))) = w;
  }
  __syncthreads();

  int wv   = tid >> 5;          // wave id = out-channel tile 0..7
  int lane = tid & 31;
  int half = lane >> 4;
  int nIdx = lane & 15;
  int o    = (wv << 4) | nIdx;

  float binit = fmode ? bias[o] : 0.0f;
  v8f acc[4];
#pragma unroll
  for (int mt = 0; mt < 4; ++mt)
    acc[mt] = v8f{binit, binit, binit, binit, binit, binit, binit, binit};

  // Per-lane window (row,col) of each M-tile's A pixel (pixel = mt*16 + lane&15).
  int prow[4], pcol[4];
#pragma unroll
  for (int mt = 0; mt < 4; ++mt) {
    int p = (mt << 4) + nIdx;                 // 0..63
    prow[mt] = p / W;
    pcol[mt] = p % W;
  }

  for (int t = 0; t < 36; ++t) {
    int dy  = t / 12;
    int rem = t - dy * 12;
    int dx  = rem >> 2;
    int c0  = (rem & 3) << 5;
    // B fragment: pre-packed, 8 contiguous dwords per lane (2x b128).
    union { unsigned int u[8]; v16bf h; } B;
    const unsigned int* bp = wp + ((((size_t)wv * 36 + t) * 32 + lane) << 3);
#pragma unroll
    for (int v = 0; v < 8; ++v) B.u[v] = bp[v];
    // 4 M-tiles reuse this B fragment.
#pragma unroll
    for (int mt = 0; mt < 4; ++mt) {
      union { unsigned int u[8]; v16bf h; } A;
      int rowbase = ((((prow[mt] + dy) * WW + (pcol[mt] + dx)) << 7) + c0);
#pragma unroll
      for (int v = 0; v < 8; ++v) {
        int koff = (half << 4) + (v << 1);
        A.u[v] = *(const unsigned int*)(sA + rowbase + koff);
      }
      acc[mt] = __builtin_amdgcn_wmma_f32_16x16x32_bf16(false, A.h, false, B.h,
                                                        (short)0, acc[mt], false, false);
    }
  }

  // Epilogue: f store / add, ReLU, stash per-pixel channel vectors in LDS.
  size_t tileBase = (size_t)(n * H + y0) * W;     // flat pixel index of (y0, 0)
#pragma unroll
  for (int mt = 0; mt < 4; ++mt) {
#pragma unroll
    for (int j = 0; j < 8; ++j) {
      int m = (mt << 4) + (half << 3) + j;        // pixel in tile, 0..63
      size_t idx = ((tileBase + m) << 7) + o;
      float s = acc[mt][j];
      if (fmode) fbuf[idx] = s;
      else       s += fbuf[idx];
      sval[(m << 7) + o] = fmaxf(s, 0.0f);
    }
  }
  __syncthreads();

  // Banded LRN (window 17, clamped) + bf16 store; 32 outputs per thread.
  for (int i = 0; i < (R * W * 128) / 256; ++i) {
    int idx = (i << 8) + tid;                     // 0..8191
    int m = idx >> 7;
    int c = idx & 127;
    const float* row = sval + (m << 7);
    float v = row[c];
    int lo = c - 8; if (lo < 0)   lo = 0;
    int hi = c + 8; if (hi > 127) hi = 127;
    float ssum = 0.0f;
    for (int j = lo; j <= hi; ++j) { float t2 = row[j]; ssum += t2 * t2; }
    float den = __powf(ssum * (0.001f / 17.0f) + 1.0f, 0.75f);
    rout[((tileBase + m) << 7) + c] = f2bf(v / den);
  }
}

// ---------------------------------------------------------------------------
// 3x3/s2/p1 maxpool on NHWC bf16 (32->16).
// ---------------------------------------------------------------------------
__global__ __launch_bounds__(256) void maxpool3s2p1_bf16(const unsigned short* __restrict__ in,
                                                         unsigned short* __restrict__ out,
                                                         int Hin) {
  int Hout = Hin >> 1;
  unsigned int gid = blockIdx.x * 256u + threadIdx.x;
  int o = gid & 127;
  unsigned int r = gid >> 7;
  int ox = r % Hout; r /= Hout;
  int oy = r % Hout;
  int n  = r / Hout;
  float m = -3.0e38f;
  for (int wy = 0; wy < 3; ++wy) {
    int iy = oy * 2 - 1 + wy;
    if ((unsigned)iy >= (unsigned)Hin) continue;
    for (int wx = 0; wx < 3; ++wx) {
      int ix = ox * 2 - 1 + wx;
      if ((unsigned)ix >= (unsigned)Hin) continue;
      m = fmaxf(m, bf2f(in[(((size_t)(n * Hin + iy) * Hin + ix) << 7) + o]));
    }
  }
  out[gid] = f2bf(m);
}

// ---------------------------------------------------------------------------
// Head: maxpool(3,2,0) on 16x16 then global max == max over [0,15)^2,
// then 128->10 MLP. One block per image.
// ---------------------------------------------------------------------------
__global__ __launch_bounds__(128) void head(const unsigned short* __restrict__ r,
                                            const float* __restrict__ mw,
                                            const float* __restrict__ mb,
                                            float* __restrict__ out) {
  __shared__ float feat[128];
  int n = blockIdx.x;
  int c = threadIdx.x;
  float m = -3.0e38f;
  for (int y = 0; y < 15; ++y)
    for (int x = 0; x < 15; ++x)
      m = fmaxf(m, bf2f(r[(((size_t)(n * 16 + y) * 16 + x) << 7) + c]));
  feat[c] = m;
  __syncthreads();
  if (c < 10) {
    float s = mb[c];
    const float* w = mw + c * 128;
    for (int k = 0; k < 128; ++k) s += feat[k] * w[k];
    out[n * 10 + c] = s;
  }
}

// ---------------------------------------------------------------------------
extern "C" void kernel_launch(void* const* d_in, const int* in_sizes, int n_in,
                              void* d_out, int out_size, void* d_ws, size_t ws_size,
                              hipStream_t stream) {
  (void)in_sizes; (void)n_in; (void)out_size; (void)ws_size;
  const float* x   = (const float*)d_in[0];
  const float* w1  = (const float*)d_in[1];
  const float* b1  = (const float*)d_in[2];
  const float* bng = (const float*)d_in[3];
  const float* bnb = (const float*)d_in[4];
  const float* fw[4] = {(const float*)d_in[5],  (const float*)d_in[8],
                        (const float*)d_in[11], (const float*)d_in[14]};
  const float* fb[4] = {(const float*)d_in[6],  (const float*)d_in[9],
                        (const float*)d_in[12], (const float*)d_in[15]};
  const float* rw[4] = {(const float*)d_in[7],  (const float*)d_in[10],
                        (const float*)d_in[13], (const float*)d_in[16]};
  const float* mw = (const float*)d_in[17];
  const float* mb = (const float*)d_in[18];
  float* out = (float*)d_out;

  // Workspace layout (peak ~163 MB):
  char* ws = (char*)d_ws;
  unsigned int* wpack = (unsigned int*)ws;                  // 8 * 294,912 B
  float* stats = (float*)(ws + 2359296);                    // 512 f32
  char* BB = ws + 2361344;
  unsigned short* x2   = (unsigned short*)(BB);                        // 32 MiB
  float*          f32b = (float*)(BB + 33554432);                      // 64 MiB
  unsigned short* rA32 = (unsigned short*)(BB + 100663296);            // 32 MiB
  unsigned short* rB32 = (unsigned short*)(BB + 134217728);            // 32 MiB
  // 16x16 stage reuses the (then dead) 64 MiB f32b region:
  float*          f16b = (float*)(BB + 33554432);                      // 16 MiB
  unsigned short* rA16 = (unsigned short*)(BB + 33554432 + 16777216);  //  8 MiB
  unsigned short* rB16 = (unsigned short*)(BB + 33554432 + 25165824);  //  8 MiB
  unsigned short* x4   = (unsigned short*)(BB + 33554432 + 33554432);  //  8 MiB

  // 1) Pack the 8 3x3 conv weights into WMMA B-fragment layout.
  const float* wsrc[8] = {fw[0], rw[0], fw[1], rw[1], fw[2], rw[2], fw[3], rw[3]};
  for (int i = 0; i < 8; ++i)
    pack_w3x3<<<288, 256, 0, stream>>>(wsrc[i], wpack + (size_t)i * 73728);

  // 2) Stem: conv1 stats -> BN coeffs -> fused conv1+BN+ReLU+pool (64->32).
  hipMemsetAsync(stats, 0, 2048, stream);
  conv1_stats<<<262144, 256, 0, stream>>>(x, w1, b1, stats);
  bn_finalize<<<1, 128, 0, stream>>>(bng, bnb, stats);
  conv1_pool<<<65536, 256, 0, stream>>>(x, w1, b1, stats, x2);

  // 3) RCL2 @32x32: f-conv then 3 recurrent convs (ping-pong rA/rB).
  conv3_wmma_lrn<32, 2><<<2048, 256, 0, stream>>>(x2,   wpack + (size_t)0 * 73728, fb[0],   f32b, rA32, 1);
  conv3_wmma_lrn<32, 2><<<2048, 256, 0, stream>>>(rA32, wpack + (size_t)1 * 73728, nullptr, f32b, rB32, 0);
  conv3_wmma_lrn<32, 2><<<2048, 256, 0, stream>>>(rB32, wpack + (size_t)1 * 73728, nullptr, f32b, rA32, 0);
  conv3_wmma_lrn<32, 2><<<2048, 256, 0, stream>>>(rA32, wpack + (size_t)1 * 73728, nullptr, f32b, rB32, 0);
  // 4) RCL3 @32x32.
  conv3_wmma_lrn<32, 2><<<2048, 256, 0, stream>>>(rB32, wpack + (size_t)2 * 73728, fb[1],   f32b, rA32, 1);
  conv3_wmma_lrn<32, 2><<<2048, 256, 0, stream>>>(rA32, wpack + (size_t)3 * 73728, nullptr, f32b, rB32, 0);
  conv3_wmma_lrn<32, 2><<<2048, 256, 0, stream>>>(rB32, wpack + (size_t)3 * 73728, nullptr, f32b, rA32, 0);
  conv3_wmma_lrn<32, 2><<<2048, 256, 0, stream>>>(rA32, wpack + (size_t)3 * 73728, nullptr, f32b, rB32, 0);
  // 5) Pool 32->16.
  maxpool3s2p1_bf16<<<16384, 256, 0, stream>>>(rB32, x4, 32);
  // 6) RCL4 @16x16.
  conv3_wmma_lrn<16, 4><<<512, 256, 0, stream>>>(x4,   wpack + (size_t)4 * 73728, fb[2],   f16b, rA16, 1);
  conv3_wmma_lrn<16, 4><<<512, 256, 0, stream>>>(rA16, wpack + (size_t)5 * 73728, nullptr, f16b, rB16, 0);
  conv3_wmma_lrn<16, 4><<<512, 256, 0, stream>>>(rB16, wpack + (size_t)5 * 73728, nullptr, f16b, rA16, 0);
  conv3_wmma_lrn<16, 4><<<512, 256, 0, stream>>>(rA16, wpack + (size_t)5 * 73728, nullptr, f16b, rB16, 0);
  // 7) RCL5 @16x16.
  conv3_wmma_lrn<16, 4><<<512, 256, 0, stream>>>(rB16, wpack + (size_t)6 * 73728, fb[3],   f16b, rA16, 1);
  conv3_wmma_lrn<16, 4><<<512, 256, 0, stream>>>(rA16, wpack + (size_t)7 * 73728, nullptr, f16b, rB16, 0);
  conv3_wmma_lrn<16, 4><<<512, 256, 0, stream>>>(rB16, wpack + (size_t)7 * 73728, nullptr, f16b, rA16, 0);
  conv3_wmma_lrn<16, 4><<<512, 256, 0, stream>>>(rA16, wpack + (size_t)7 * 73728, nullptr, f16b, rB16, 0);
  // 8) Head: pool(3,2,0) + global max + MLP.
  head<<<128, 128, 0, stream>>>(rB16, mw, mb, out);
}
